// Feedback_11141145166388
// MI455X (gfx1250) — matrix operand order repaired and checked by reading, not compiled, not run on hardware
//
#include <hip/hip_runtime.h>
#include <hip/hip_bf16.h>

typedef __attribute__((ext_vector_type(16))) __bf16        v16bf;
typedef __attribute__((ext_vector_type(8)))  float         v8f;
typedef __attribute__((ext_vector_type(4)))  unsigned int  u32x4;

#define HCDIM 1024
#define BATCH 256
#define TIMET 192
#define ZCOLS 4096

union ABfrag { v16bf v; u32x4 q[2]; };

// ---------------------------------------------------------------------------
// One-time: convert Wh (fp32, 1024 x 4096 row-major) to bf16, repacked into
// WMMA-B fragment order. Fragment (kt, ct) covers K = kt*32..+31, N = ct*16..+15.
// B-operand per-lane layout (16x16x32 bf16): lane l: n = l&15, khalf = l>>4;
// element e (0..15): k = (e>>3)*16 + khalf*8 + (e&7). Each lane's 16 bf16 (32 B) contiguous:
//   Wp[ ((kt*256 + ct)*32 + lane)*16 + e ]
// ---------------------------------------------------------------------------
__global__ __launch_bounds__(256) void pack_wh_kernel(const float* __restrict__ Wh,
                                                      __bf16* __restrict__ Wp) {
    int idx  = blockIdx.x * 256 + threadIdx.x;        // 0 .. 4M-1
    int e    = idx & 15;
    int lane = (idx >> 4) & 31;
    int ct   = (idx >> 9) & 255;
    int kt   = idx >> 17;
    int k    = kt * 32 + (e >> 3) * 16 + (lane >> 4) * 8 + (e & 7);
    int zc   = ct * 16 + (lane & 15);
    Wp[idx]  = (__bf16)Wh[(size_t)k * ZCOLS + zc];
}

__global__ __launch_bounds__(256) void init_state_kernel(__bf16* __restrict__ h0,
                                                         float* __restrict__ c) {
    int i = blockIdx.x * 256 + threadIdx.x;           // 0 .. 256*1024-1
    h0[i] = (__bf16)0.0f;
    c[i]  = 0.0f;
}

__device__ __forceinline__ float sigmoidf_dev(float x) {
    return 1.0f / (1.0f + __expf(-x));
}

// Overflow-safe fast tanh (avoids the heavyweight OCML tanhf path).
__device__ __forceinline__ float tanhf_fast(float x) {
    float e = __expf(-2.0f * fabsf(x));
    float r = (1.0f - e) / (1.0f + e);
    return copysignf(r, x);
}

// Load one k-tile's A fragment + 4 gate B fragments.
__device__ __forceinline__ void load_frags(ABfrag& a, ABfrag b[4],
        const __bf16* __restrict__ ap, const __bf16* __restrict__ Wp,
        const size_t boff[4], int kt) {
    const __bf16* apk = ap + kt * 32;
    a.q[0] = *(const u32x4*)(apk);
    a.q[1] = *(const u32x4*)(apk + 16);
    size_t kbase = (size_t)kt << 17;                  // kt * 256*32*16
#pragma unroll
    for (int g = 0; g < 4; ++g) {
        const __bf16* bp = Wp + kbase + boff[g];
        b[g].q[0] = *(const u32x4*)(bp);
        b[g].q[1] = *(const u32x4*)(bp + 8);
    }
}

__device__ __forceinline__ void do_mma(v8f acc[4], const ABfrag& a, const ABfrag b[4]) {
#pragma unroll
    for (int g = 0; g < 4; ++g) {
        acc[g] = __builtin_amdgcn_wmma_f32_16x16x32_bf16(
            false, a.v, false, b[g].v, (short)0, acc[g], false, false);
    }
}

// ---------------------------------------------------------------------------
// One LSTM timestep:  z = h @ Wh + x @ Wx + b, gates, update c (in place) and
// h (ping-pong hin -> hout).  Grid (4, 32), block 256 (8 waves).
// Wave w: row block r = w>>1 (16 rows), col tile ch = w&1 (16 h-cols),
// computing that tile for all 4 gates (z cols g*1024 + hcol).
// mode 0: x = inputs[:, t, :] with -1 mask; mode 1: x = pred feedback.
// Inner K loop: fully unrolled, 3-deep fragment pipeline (prefetch distance
// 2 k-tiles) so WMMA waits never have to drain the load queue to zero.
// ---------------------------------------------------------------------------
__global__ __launch_bounds__(256) void lstm_step_kernel(
        const __bf16* __restrict__ Wp,
        const float*  __restrict__ xin,
        const float*  __restrict__ Wx,     // (2, 4096)
        const float*  __restrict__ bvec,   // (4096,)
        const __bf16* __restrict__ hin,
        __bf16*       __restrict__ hout,
        float*        __restrict__ cbuf,
        int t, int mode) {
    int tid   = threadIdx.x;
    int wave  = tid >> 5;
    int lane  = tid & 31;
    int r     = wave >> 1;
    int chv   = wave & 1;
    int rowBase = blockIdx.x * 64 + r * 16;
    int colBase = blockIdx.y * 32 + chv * 16;
    int m     = lane & 15;
    int khalf = lane >> 4;

    v8f zero = {0.f, 0.f, 0.f, 0.f, 0.f, 0.f, 0.f, 0.f};
    v8f acc[4] = {zero, zero, zero, zero};

    // A fragment base: 16-bit A 16x32 layout — lane (m, khalf) needs
    // K = kb + khalf*8 + {0..7}  and  K = kb + 16 + khalf*8 + {0..7}
    const __bf16* ap = hin + (size_t)(rowBase + m) * HCDIM + khalf * 8;

    size_t boff[4];
#pragma unroll
    for (int g = 0; g < 4; ++g) {
        boff[g] = ((size_t)((g * HCDIM + colBase) >> 4) * 32 + lane) * 16;
    }

    // 3-stage software pipeline over 32 k-tiles.
    ABfrag aS[3];
    ABfrag bS[3][4];
    load_frags(aS[0], bS[0], ap, Wp, boff, 0);
    load_frags(aS[1], bS[1], ap, Wp, boff, 1);
#pragma unroll
    for (int kt = 0; kt < 32; ++kt) {
        int cur = kt % 3;
        int nxt = (kt + 2) % 3;
        if (kt + 2 < 32) {
            load_frags(aS[nxt], bS[nxt], ap, Wp, boff, kt + 2);
        }
        do_mma(acc, aS[cur], bS[cur]);
    }

    // Epilogue: C/D layout — lane l, vgpr v: M = v + 8*(l>>4), N = l&15
    int n    = lane & 15;
    int hcol = colBase + n;
    float wx0[4], wx1[4], bb[4];
#pragma unroll
    for (int g = 0; g < 4; ++g) {
        int zc = g * HCDIM + hcol;
        wx0[g] = Wx[zc];
        wx1[g] = Wx[ZCOLS + zc];
        bb[g]  = bvec[zc];
    }
#pragma unroll
    for (int v = 0; v < 8; ++v) {
        int row = rowBase + khalf * 8 + v;
        float x0, x1;
        bool live = true;
        if (mode == 0) {
            const float* xp = xin + ((size_t)row * TIMET + t) * 2;
            x0 = xp[0];
            x1 = xp[1];
            live = (x0 != -1.0f) || (x1 != -1.0f);
        } else {
            x0 = xin[row * 2 + 0];
            x1 = xin[row * 2 + 1];
        }
        float iz = acc[0][v] + x0 * wx0[0] + x1 * wx1[0] + bb[0];
        float fz = acc[1][v] + x0 * wx0[1] + x1 * wx1[1] + bb[1];
        float gz = acc[2][v] + x0 * wx0[2] + x1 * wx1[2] + bb[2];
        float oz = acc[3][v] + x0 * wx0[3] + x1 * wx1[3] + bb[3];
        float ig = sigmoidf_dev(iz);
        float fg = sigmoidf_dev(fz);
        float gg = tanhf_fast(gz);
        float og = sigmoidf_dev(oz);
        size_t idx = (size_t)row * HCDIM + hcol;
        float cold = cbuf[idx];
        float cnew = fg * cold + ig * gg;
        float hnew = og * tanhf_fast(cnew);
        if (live) {
            cbuf[idx] = cnew;
            hout[idx] = (__bf16)hnew;
        } else {
            hout[idx] = hin[idx];   // carry old h through ping-pong
        }
    }
}

// ---------------------------------------------------------------------------
// pred = h @ Wd + bd  (256x1024 . 1024x2). One block per batch row.
// Writes d_out[row, step, :] and the feedback buffer.
// ---------------------------------------------------------------------------
__global__ __launch_bounds__(256) void pred_step_kernel(
        const __bf16* __restrict__ h,
        const float*  __restrict__ Wd,   // (1024, 2)
        const float*  __restrict__ bd,   // (2,)
        float*        __restrict__ pred,
        float*        __restrict__ out,
        int step) {
    __shared__ float red0[256];
    __shared__ float red1[256];
    int row = blockIdx.x;
    int tid = threadIdx.x;
    float p0 = 0.f, p1 = 0.f;
    const __bf16* hp = h + (size_t)row * HCDIM;
    for (int k = tid; k < HCDIM; k += 256) {
        float hv = (float)hp[k];
        p0 += hv * Wd[k * 2 + 0];
        p1 += hv * Wd[k * 2 + 1];
    }
    red0[tid] = p0;
    red1[tid] = p1;
    __syncthreads();
    for (int s = 128; s > 0; s >>= 1) {
        if (tid < s) {
            red0[tid] += red0[tid + s];
            red1[tid] += red1[tid + s];
        }
        __syncthreads();
    }
    if (tid == 0) {
        float q0 = red0[0] + bd[0];
        float q1 = red1[0] + bd[1];
        pred[row * 2 + 0] = q0;
        pred[row * 2 + 1] = q1;
        out[((size_t)row * 128 + step) * 2 + 0] = q0;
        out[((size_t)row * 128 + step) * 2 + 1] = q1;
    }
}

extern "C" void kernel_launch(void* const* d_in, const int* in_sizes, int n_in,
                              void* d_out, int out_size, void* d_ws, size_t ws_size,
                              hipStream_t stream) {
    const float* inputs = (const float*)d_in[0];  // (256, 192, 2)
    const float* Wx     = (const float*)d_in[1];  // (2, 4096)
    const float* Wh     = (const float*)d_in[2];  // (1024, 4096)
    const float* bvec   = (const float*)d_in[3];  // (4096,)
    const float* Wd     = (const float*)d_in[4];  // (1024, 2)
    const float* bd     = (const float*)d_in[5];  // (2,)
    float* out = (float*)d_out;

    char* ws = (char*)d_ws;
    __bf16* Wp   = (__bf16*)ws;                                  // 8 MiB
    __bf16* h0   = (__bf16*)(ws + (size_t)8 * 1024 * 1024);      // 512 KiB
    __bf16* h1   = h0 + (size_t)BATCH * HCDIM;                   // 512 KiB
    float*  cbuf = (float*)(ws + (size_t)9 * 1024 * 1024 + 512 * 1024); // 1 MiB
    float*  pred = (float*)(ws + (size_t)10 * 1024 * 1024 + 512 * 1024);

    // 1) pack Wh -> bf16 WMMA fragments  (1024*4096 elements)
    pack_wh_kernel<<<16384, 256, 0, stream>>>(Wh, Wp);
    // 2) zero h0, c  (256*1024 elements each)
    init_state_kernel<<<1024, 256, 0, stream>>>(h0, cbuf);

    __bf16* hin  = h0;
    __bf16* hout = h1;
    dim3 sgrid(BATCH / 64, HCDIM / 32);

    // 3) warm-up: 64 masked steps
    for (int t = 0; t < 64; ++t) {
        lstm_step_kernel<<<sgrid, 256, 0, stream>>>(
            Wp, inputs, Wx, bvec, hin, hout, cbuf, t, 0);
        __bf16* tmp = hin; hin = hout; hout = tmp;
    }
    // 4) pred0
    pred_step_kernel<<<BATCH, 256, 0, stream>>>(hin, Wd, bd, pred, out, 0);
    // 5) decode: 127 feedback steps
    for (int s = 1; s <= 127; ++s) {
        lstm_step_kernel<<<sgrid, 256, 0, stream>>>(
            Wp, pred, Wx, bvec, hin, hout, cbuf, 0, 1);
        __bf16* tmp = hin; hin = hout; hout = tmp;
        pred_step_kernel<<<BATCH, 256, 0, stream>>>(hin, Wd, bd, pred, out, s);
    }
}
